// ChebGraphConv_59811714564812
// MI455X (gfx1250) — matrix operand order repaired and checked by reading, not compiled, not run on hardware
//
#include <hip/hip_runtime.h>

// ---------------------------------------------------------------------------
// ChebGraphConv on MI455X (gfx1250, wave32, WMMA)
//
// out[b,t,n,o] = sum_k diag(Tk)[n] * (x[b,t,n,:] @ Theta_k)[o] + sum_k bias[k,o]
//
// 3 GEMMs sharing the A operand (x), v_wmma_f32_16x16x32_bf16.
// Persistent waves: each wave keeps all 24 packed Theta B-fragments in VGPRs
// and streams 8 consecutive 16-row tiles of x through them.
// f32->bf16 conversion via v_cvt_pk_bf16_f32 when available.
// ---------------------------------------------------------------------------

typedef __attribute__((ext_vector_type(16))) __bf16 v16bf;
typedef __attribute__((ext_vector_type(2)))  __bf16 v2bf;
typedef __attribute__((ext_vector_type(8)))  float  v8f;

#define N_NODES   1024
#define C_IN      64
#define C_OUT     64
#define BT        768                  // B*T = 32*24
#define M_ROWS    (BT * N_NODES)       // 786432
#define TILES     (M_ROWS / 16)        // 49152
#define WPB       8                    // waves per block (256 threads)
#define TPW       8                    // tiles per wave (persistent loop)

// workspace layout (bytes)
#define WPACK_OFF 0                    // ushort[3*2*4*32*16] = 24576 B (bf16 Theta, B-layout)
#define D_OFF     65536                // float[3*1024]      = 12288 B (diagonals)
#define BSUM_OFF  (65536 + 12288)      // float[64]          (sum_k bias[k])

__device__ __forceinline__ unsigned short f2bf(float f) {
    union { float f; unsigned u; } v; v.f = f;
    unsigned r = v.u + 0x7FFFu + ((v.u >> 16) & 1u);   // round-to-nearest-even
    return (unsigned short)(r >> 16);
}

// packed 2x f32 -> 2x bf16 (v_cvt_pk_bf16_f32 on CDNA5)
__device__ __forceinline__ v2bf pk_bf16(float lo, float hi) {
#if __has_builtin(__builtin_amdgcn_cvt_pk_bf16_f32)
    auto p = __builtin_amdgcn_cvt_pk_bf16_f32(lo, hi);
    return __builtin_bit_cast(v2bf, p);
#else
    v2bf r; r[0] = (__bf16)lo; r[1] = (__bf16)hi; return r;
#endif
}

union V16 { v16bf v; v2bf p[8]; unsigned short u[16]; };

// ---------------------------------------------------------------------------
// prep: extract diag(Tk), sum bias, and pack Theta_k (f32) into a contiguous-k
// 16-bit B-matrix (32x16) VGPR layout as bf16:
//   slot = ((k*2 + kstep)*4 + otile)*32 + lane, 16 bf16 per slot
//   element j of lane L holds Theta[k][ 32*kstep + (L<16?0:16) + j ][ 16*otile + L%16 ]
// ---------------------------------------------------------------------------
__global__ __launch_bounds__(256) void cheb_prep(
    const float* __restrict__ Tks, const float* __restrict__ Theta,
    const float* __restrict__ bias,
    unsigned short* __restrict__ wpack, float* __restrict__ dA,
    float* __restrict__ bsum)
{
    const int tid = threadIdx.x;
    const int bid = blockIdx.x;

    if (bid < 12) {
        // diagonals: 3*1024 elements, one per thread
        int idx = bid * 256 + tid;                  // 0..3071
        int k = idx >> 10, n = idx & (N_NODES - 1);
        dA[idx] = Tks[(size_t)k * N_NODES * N_NODES + (size_t)n * (N_NODES + 1)];
    } else {
        // Theta pack: 3*2*4*32 = 768 slots of 16 bf16
        for (int s = tid; s < 768; s += 256) {
            int k     = s >> 8;
            int r     = s & 255;
            int kstep = r >> 7;
            int r2    = r & 127;
            int ot    = r2 >> 5;
            int lane  = r2 & 31;
            int cbase = kstep * 32 + ((lane & 16) ? 16 : 0);
            int o     = ot * 16 + (lane & 15);
            const float* th = Theta + (size_t)k * C_IN * C_OUT;
            unsigned short* wp = wpack + (size_t)s * 16;
            #pragma unroll
            for (int j = 0; j < 16; j++)
                wp[j] = f2bf(th[(cbase + j) * C_OUT + o]);
        }
        if (tid < C_OUT)
            bsum[tid] = bias[tid] + bias[C_OUT + tid] + bias[2 * C_OUT + tid];
    }
}

// ---------------------------------------------------------------------------
// main GEMM: persistent wave, TPW consecutive 16x64 tiles of x.
// Per tile: 2 K-steps x 3 Theta_k x 4 O-tiles = 24 v_wmma_f32_16x16x32_bf16.
// ---------------------------------------------------------------------------
__global__ __launch_bounds__(256) void cheb_wmma(
    const float* __restrict__ x, const unsigned short* __restrict__ wpack,
    const float* __restrict__ dA, const float* __restrict__ bsum,
    float* __restrict__ out)
{
    const int lane = threadIdx.x & 31;
    const int wave = threadIdx.x >> 5;
    const int hi   = lane >> 4;        // 0 for lanes 0-15, 1 for 16-31
    const int mlo  = lane & 15;
    const int tile0 = (blockIdx.x * WPB + wave) * TPW;

    // ---- load all 24 pre-packed bf16 B fragments once (32B read per lane) ----
    v16bf bm[3][2][4];
    #pragma unroll
    for (int k = 0; k < 3; k++)
        #pragma unroll
        for (int ks = 0; ks < 2; ks++)
            #pragma unroll
            for (int ot = 0; ot < 4; ot++) {
                int slot = ((k * 2 + ks) * 4 + ot) * 32 + lane;
                bm[k][ks][ot] = *reinterpret_cast<const v16bf*>(wpack + (size_t)slot * 16);
            }

    // summed bias for this lane's output columns (constant across tiles)
    float bs[4];
    #pragma unroll
    for (int ot = 0; ot < 4; ot++) bs[ot] = bsum[ot * 16 + mlo];

    for (int t = 0; t < TPW; ++t) {
        const int tile = tile0 + t;
        // A layout (16-bit 16x32): lane L -> row M = L%16; element j holds
        // K = 32*kstep + (L<16?0:16) + j
        const float* xrow = x + ((size_t)tile * 16 + mlo) * C_IN;

        // prefetch next tile's slice of x (global_prefetch_b8)
        __builtin_prefetch(xrow + 16 * C_IN, 0, 1);

        v8f acc[3][4] = {};

        #pragma unroll
        for (int ks = 0; ks < 2; ks++) {
            const int c0 = ks * 32 + hi * 16;
            float4 f0 = *reinterpret_cast<const float4*>(xrow + c0);
            float4 f1 = *reinterpret_cast<const float4*>(xrow + c0 + 4);
            float4 f2 = *reinterpret_cast<const float4*>(xrow + c0 + 8);
            float4 f3 = *reinterpret_cast<const float4*>(xrow + c0 + 12);

            V16 a;
            a.p[0] = pk_bf16(f0.x, f0.y);  a.p[1] = pk_bf16(f0.z, f0.w);
            a.p[2] = pk_bf16(f1.x, f1.y);  a.p[3] = pk_bf16(f1.z, f1.w);
            a.p[4] = pk_bf16(f2.x, f2.y);  a.p[5] = pk_bf16(f2.z, f2.w);
            a.p[6] = pk_bf16(f3.x, f3.y);  a.p[7] = pk_bf16(f3.z, f3.w);

            #pragma unroll
            for (int k = 0; k < 3; k++)
                #pragma unroll
                for (int ot = 0; ot < 4; ot++)
                    acc[k][ot] = __builtin_amdgcn_wmma_f32_16x16x32_bf16(
                        /*neg_a=*/false, a.v, /*neg_b=*/false, bm[k][ks][ot],
                        /*c_mod=*/(short)0, acc[k][ot],
                        /*reuse_a=*/false, /*reuse_b=*/false);
        }

        // ---- epilogue: per-node diagonal scales + summed bias, f32 stores ----
        // C/D layout: VGPR r, lanes 0-15 -> M=r, lanes 16-31 -> M=r+8; col = lane%16
        #pragma unroll
        for (int r = 0; r < 8; r++) {
            size_t g  = (size_t)tile * 16 + r + hi * 8;     // global row
            int    n  = (int)(g & (N_NODES - 1));           // node id
            float  d0 = dA[n];
            float  d1 = dA[N_NODES + n];
            float  d2 = dA[2 * N_NODES + n];
            #pragma unroll
            for (int ot = 0; ot < 4; ot++) {
                float v = d0 * acc[0][ot][r] + d1 * acc[1][ot][r] + d2 * acc[2][ot][r] + bs[ot];
                out[g * C_OUT + ot * 16 + mlo] = v;
            }
        }
    }
}

extern "C" void kernel_launch(void* const* d_in, const int* in_sizes, int n_in,
                              void* d_out, int out_size, void* d_ws, size_t ws_size,
                              hipStream_t stream) {
    const float* x     = (const float*)d_in[0];  // [32,24,1024,64]
    const float* Tks   = (const float*)d_in[1];  // [3,1024,1024]
    const float* Theta = (const float*)d_in[2];  // [3,64,64]
    const float* bias  = (const float*)d_in[3];  // [3,64]
    float*       out   = (float*)d_out;          // [32,24,1024,64]

    unsigned short* wpack = (unsigned short*)((char*)d_ws + WPACK_OFF);
    float*          dA    = (float*)((char*)d_ws + D_OFF);
    float*          bsum  = (float*)((char*)d_ws + BSUM_OFF);

    // block 0-11: diagonals; block 12: Theta pack + bias sum
    cheb_prep<<<13, 256, 0, stream>>>(Tks, Theta, bias, wpack, dA, bsum);

    // 49152 tiles / (8 waves * 8 tiles-per-wave) = 768 blocks
    cheb_wmma<<<TILES / (WPB * TPW), 256, 0, stream>>>(x, wpack, dA, bsum, out);
}
